// GATModel_80264348828259
// MI455X (gfx1250) — matrix operand order repaired and checked by reading, not compiled, not run on hardware
//
#include <hip/hip_runtime.h>
#include <hip/hip_bf16.h>

#define HEADS 4
#define NEG_SLOPE 0.2f

typedef __attribute__((ext_vector_type(16))) __bf16 v16bf;
typedef __attribute__((ext_vector_type(8)))  float  v8f;

static constexpr int N_X   = 9000;
static constexpr int N_C   = 1000;
static constexpr int NTOT  = 10000;
static constexpr int E_IN  = 40000;
static constexpr int E_TOT = E_IN + NTOT;   // self loops appended
static constexpr int D_IN0 = 400;
static constexpr int D_H1  = 512;
static constexpr int D_EMB = 1024;
static constexpr int D_GAT = HEADS * 1024;  // 4096
static constexpr int D_OUT = 460;

static constexpr int KB0 = (D_IN0 + 31) / 32;  // 13 (K=400, zero padded)
static constexpr int KB1 = D_H1  / 32;         // 16
static constexpr int KB2 = D_EMB / 32;         // 32

// ---------------------------------------------------------------------------
// Operand packing into wave32 WMMA fragments (contiguous k = 16h + i, not the ISA table order), f32 -> bf16.
// A-pack: P[(m*KB + kb)*32 + h*16 + i] = A[m][kb*32 + h*16 + i]
//   -> each half-wave's 16 A-fragment elements are contiguous (32B).
// B-pack: P[((kb*N) + n)*32 + j]       = B[kb*32 + j][n]
//   -> each half-wave's 16 B-fragment elements are contiguous (32B).
// K is zero-padded to KB*32, so the GEMM needs no K-tail path.
// ---------------------------------------------------------------------------
__global__ void pack_A_kernel(const float* __restrict__ A, __bf16* __restrict__ P,
                              int M, int K, int KB) {
  const size_t idx = (size_t)blockIdx.x * blockDim.x + threadIdx.x;
  const size_t total = (size_t)M * KB * 32;
  if (idx >= total) return;
  const int j  = (int)(idx & 31);
  const size_t mk = idx >> 5;
  const int kb = (int)(mk % KB);
  const int m  = (int)(mk / KB);
  const int h = j >> 4, i = j & 15;
  const int k = kb * 32 + h * 16 + i;
  P[idx] = (__bf16)((k < K) ? A[(size_t)m * K + k] : 0.0f);
}

__global__ void pack_B_kernel(const float* __restrict__ B, __bf16* __restrict__ P,
                              int K, int N, int KB) {
  const size_t idx = (size_t)blockIdx.x * blockDim.x + threadIdx.x;
  const size_t total = (size_t)KB * N * 32;
  if (idx >= total) return;
  const int j = (int)(idx & 31);
  const size_t kn = idx >> 5;
  const int n  = (int)(kn % N);
  const int kb = (int)(kn / N);
  const int k = kb * 32 + j;
  P[idx] = (__bf16)((k < K) ? B[(size_t)k * N + n] : 0.0f);
}

// ---------------------------------------------------------------------------
// GEMM on packed bf16 operands: C[M,N] = act(A@B + bias), f32 accumulate.
// Block = 8 waves; wave tile = 32(M) x 64(N) -> block 256 x 64.
// Inner loop: 6 x 32B fragment loads (global_load_b128 x12) + 8 WMMAs.
// Ragged M/N handled by index clamping (garbage lanes never stored).
// ---------------------------------------------------------------------------
template <int N, bool RELU>
__global__ __launch_bounds__(256)
void gemm_wmma_packed(const __bf16* __restrict__ Ap, const __bf16* __restrict__ Bp,
                      const float* __restrict__ bias, float* __restrict__ C,
                      int M, int KB) {
  const int lane = threadIdx.x & 31;
  const int wave = threadIdx.x >> 5;
  const int row0 = blockIdx.y * 256 + wave * 32;
  const int col0 = blockIdx.x * 64;
  if (row0 >= M) return;  // uniform per wave: EXEC stays all-ones for WMMA

  const int half = lane >> 4;
  const int l15  = lane & 15;

  // fragment base pointers (rows/cols clamped; half-offset folded in)
  // NOTE: named scalars (not an array) so clang keeps the global addrspace
  // and emits global_load_b128 (LOADcnt only), not flat_load.
  const __bf16* a0p = Ap + ((size_t)min(row0 + l15,      M - 1) * KB) * 32 + half * 16;
  const __bf16* a1p = Ap + ((size_t)min(row0 + 16 + l15, M - 1) * KB) * 32 + half * 16;
  const __bf16* bp0 = Bp + (size_t)min(col0 +  0 + l15, N - 1) * 32 + half * 16;
  const __bf16* bp1 = Bp + (size_t)min(col0 + 16 + l15, N - 1) * 32 + half * 16;
  const __bf16* bp2 = Bp + (size_t)min(col0 + 32 + l15, N - 1) * 32 + half * 16;
  const __bf16* bp3 = Bp + (size_t)min(col0 + 48 + l15, N - 1) * 32 + half * 16;

  v8f acc[2][4];
#pragma unroll
  for (int mi = 0; mi < 2; ++mi)
#pragma unroll
    for (int t = 0; t < 4; ++t)
#pragma unroll
      for (int r = 0; r < 8; ++r) acc[mi][t][r] = 0.0f;

  for (int kb = 0; kb < KB; ++kb) {
    __builtin_prefetch(a0p + 64, 0, 1);  // A chunk two k-blocks ahead; speculative
    const v16bf a0 = *(const v16bf*)a0p;
    const v16bf a1 = *(const v16bf*)a1p;
    const v16bf b0 = *(const v16bf*)bp0;
    const v16bf b1 = *(const v16bf*)bp1;
    const v16bf b2 = *(const v16bf*)bp2;
    const v16bf b3 = *(const v16bf*)bp3;
    acc[0][0] = __builtin_amdgcn_wmma_f32_16x16x32_bf16(
        false, a0, false, b0, (short)0, acc[0][0], false, false);
    acc[1][0] = __builtin_amdgcn_wmma_f32_16x16x32_bf16(
        false, a1, false, b0, (short)0, acc[1][0], false, false);
    acc[0][1] = __builtin_amdgcn_wmma_f32_16x16x32_bf16(
        false, a0, false, b1, (short)0, acc[0][1], false, false);
    acc[1][1] = __builtin_amdgcn_wmma_f32_16x16x32_bf16(
        false, a1, false, b1, (short)0, acc[1][1], false, false);
    acc[0][2] = __builtin_amdgcn_wmma_f32_16x16x32_bf16(
        false, a0, false, b2, (short)0, acc[0][2], false, false);
    acc[1][2] = __builtin_amdgcn_wmma_f32_16x16x32_bf16(
        false, a1, false, b2, (short)0, acc[1][2], false, false);
    acc[0][3] = __builtin_amdgcn_wmma_f32_16x16x32_bf16(
        false, a0, false, b3, (short)0, acc[0][3], false, false);
    acc[1][3] = __builtin_amdgcn_wmma_f32_16x16x32_bf16(
        false, a1, false, b3, (short)0, acc[1][3], false, false);
    a0p += 32; a1p += 32;
    bp0 += (size_t)N * 32; bp1 += (size_t)N * 32;
    bp2 += (size_t)N * 32; bp3 += (size_t)N * 32;
  }

  // C/D layout: VGPR r -> M = r + 8*(lane>=16), N = lane%16 within 16x16 tile.
  const int rbase = half * 8;
  float bcol[4];
#pragma unroll
  for (int t = 0; t < 4; ++t) bcol[t] = bias[min(col0 + t * 16 + l15, N - 1)];

  if (row0 + 32 <= M && col0 + 64 <= N) {  // interior: unguarded stores
#pragma unroll
    for (int mi = 0; mi < 2; ++mi)
#pragma unroll
      for (int t = 0; t < 4; ++t) {
        float* Cp = C + (size_t)(row0 + mi * 16 + rbase) * (size_t)N
                      + (col0 + t * 16 + l15);
#pragma unroll
        for (int r = 0; r < 8; ++r) {
          float v = acc[mi][t][r] + bcol[t];
          if (RELU) v = fmaxf(v, 0.0f);
          Cp[(size_t)r * N] = v;
        }
      }
  } else {  // boundary tile
#pragma unroll
    for (int mi = 0; mi < 2; ++mi)
#pragma unroll
      for (int t = 0; t < 4; ++t)
#pragma unroll
        for (int r = 0; r < 8; ++r) {
          const int row = row0 + mi * 16 + rbase + r;
          const int col = col0 + t * 16 + l15;
          if (row < M && col < N) {
            float v = acc[mi][t][r] + bcol[t];
            if (RELU) v = fmaxf(v, 0.0f);
            C[(size_t)row * N + col] = v;
          }
        }
  }
}

// ---------------------------------------------------------------------------
// Init: agg = 0; m = -inf; z = 0
// ---------------------------------------------------------------------------
__global__ void init_kernel(float* __restrict__ agg, float* __restrict__ m,
                            float* __restrict__ z, int n_agg, int n_mz) {
  int i = blockIdx.x * blockDim.x + threadIdx.x;
  if (i < n_agg) agg[i] = 0.0f;
  if (i < n_mz) { m[i] = -3.0e38f; z[i] = 0.0f; }
}

__device__ __forceinline__ void edge_sd(const int* __restrict__ edges, int e,
                                        int& src, int& dst) {
  if (e < E_IN) { src = edges[2 * e]; dst = edges[2 * e + 1]; }
  else          { src = dst = e - E_IN; }
}

// ---------------------------------------------------------------------------
// alpha[e,h] = dot(leaky_relu(XL[src,h,:] + XR[dst,h,:]), att[h,:])
// One wave per (edge, head); float4 channel loop + wave32 shuffle reduction.
// ---------------------------------------------------------------------------
__global__ __launch_bounds__(256)
void alpha_kernel(const float* __restrict__ XL, const float* __restrict__ XR,
                  const int* __restrict__ edges, const float* __restrict__ att,
                  float* __restrict__ alpha) {
  const int gid  = blockIdx.x * (blockDim.x >> 5) + (threadIdx.x >> 5);
  const int lane = threadIdx.x & 31;
  if (gid >= E_TOT * HEADS) return;
  const int e = gid >> 2, h = gid & 3;
  int src, dst; edge_sd(edges, e, src, dst);
  const float* pl = XL + (size_t)src * D_GAT + h * 1024;
  const float* pr = XR + (size_t)dst * D_GAT + h * 1024;
  const float* pa = att + h * 1024;
  float s = 0.0f;
#pragma unroll
  for (int j = 0; j < 8; ++j) {
    const int c = j * 128 + lane * 4;
    const float4 l = *(const float4*)(pl + c);
    const float4 r = *(const float4*)(pr + c);
    const float4 a = *(const float4*)(pa + c);
    float v;
    v = l.x + r.x; s += ((v > 0.f) ? v : NEG_SLOPE * v) * a.x;
    v = l.y + r.y; s += ((v > 0.f) ? v : NEG_SLOPE * v) * a.y;
    v = l.z + r.z; s += ((v > 0.f) ? v : NEG_SLOPE * v) * a.z;
    v = l.w + r.w; s += ((v > 0.f) ? v : NEG_SLOPE * v) * a.w;
  }
#pragma unroll
  for (int off = 16; off > 0; off >>= 1) s += __shfl_xor(s, off, 32);
  if (lane == 0) alpha[gid] = s;
}

// segment max over dst (float atomicMax -> global_atomic_max_num_f32)
__global__ void segmax_kernel(const float* __restrict__ alpha,
                              const int* __restrict__ edges, float* __restrict__ m) {
  int gid = blockIdx.x * blockDim.x + threadIdx.x;
  if (gid >= E_TOT * HEADS) return;
  const int e = gid >> 2, h = gid & 3;
  int src, dst; edge_sd(edges, e, src, dst);
  (void)src;
  atomicMax(&m[dst * HEADS + h], alpha[gid]);
}

// a = exp(alpha - m[dst]); z[dst] += a (in place over alpha)
__global__ void expsum_kernel(float* __restrict__ alpha, const int* __restrict__ edges,
                              const float* __restrict__ m, float* __restrict__ z) {
  int gid = blockIdx.x * blockDim.x + threadIdx.x;
  if (gid >= E_TOT * HEADS) return;
  const int e = gid >> 2, h = gid & 3;
  int src, dst; edge_sd(edges, e, src, dst);
  (void)src;
  float a = __expf(alpha[gid] - m[dst * HEADS + h]);
  alpha[gid] = a;
  atomicAdd(&z[dst * HEADS + h], a);
}

// agg[dst, c] += sum_h 0.25 * a[e,h]/(z[dst,h]+eps) * XL[src, h*1024 + c]
// One block (256 threads) per edge; each thread handles 4 contiguous channels.
__global__ __launch_bounds__(256)
void agg_kernel(const float* __restrict__ XL, const float* __restrict__ alpha,
                const float* __restrict__ z, const int* __restrict__ edges,
                float* __restrict__ agg) {
  const int e = blockIdx.x;
  int src, dst; edge_sd(edges, e, src, dst);
  float coef[HEADS];
#pragma unroll
  for (int h = 0; h < HEADS; ++h)
    coef[h] = 0.25f * alpha[e * HEADS + h] / (z[dst * HEADS + h] + 1e-16f);
  const int c = threadIdx.x * 4;
  float s0 = 0.f, s1 = 0.f, s2 = 0.f, s3 = 0.f;
#pragma unroll
  for (int h = 0; h < HEADS; ++h) {
    const float4 v = *(const float4*)(XL + (size_t)src * D_GAT + h * 1024 + c);
    s0 += coef[h] * v.x; s1 += coef[h] * v.y;
    s2 += coef[h] * v.z; s3 += coef[h] * v.w;
  }
  float* p = agg + (size_t)dst * 1024 + c;
  atomicAdd(p + 0, s0); atomicAdd(p + 1, s1);
  atomicAdd(p + 2, s2); atomicAdd(p + 3, s3);
}

__global__ void bias_add_kernel(float* __restrict__ agg, const float* __restrict__ bias,
                                int n) {
  int i = blockIdx.x * blockDim.x + threadIdx.x;
  if (i < n) agg[i] += bias[i & 1023];
}

// ---------------------------------------------------------------------------
extern "C" void kernel_launch(void* const* d_in, const int* in_sizes, int n_in,
                              void* d_out, int out_size, void* d_ws, size_t ws_size,
                              hipStream_t stream) {
  const float* x    = (const float*)d_in[0];
  const float* emb  = (const float*)d_in[1];
  const int*   eidx = (const int*)d_in[2];
  const float* exps = (const float*)d_in[3];
  const float* W1   = (const float*)d_in[4];
  const float* b1   = (const float*)d_in[5];
  const float* W2   = (const float*)d_in[6];
  const float* b2   = (const float*)d_in[7];
  const float* Wl   = (const float*)d_in[8];
  const float* bl   = (const float*)d_in[9];
  const float* Wr   = (const float*)d_in[10];
  const float* br   = (const float*)d_in[11];
  const float* att  = (const float*)d_in[12];
  const float* gatb = (const float*)d_in[13];
  const float* Wfc  = (const float*)d_in[14];
  const float* bfc  = (const float*)d_in[15];
  float* out = (float*)d_out;

  // ---- workspace layout: f32 region, then bf16 packed region (64B aligned) --
  float* ws = (float*)d_ws;
  size_t off = 0;
  float* xc    = ws + off; off += (size_t)NTOT * D_EMB;      // 10.24M
  float* H1    = ws + off; off += (size_t)N_X * D_H1;        //  4.61M
  float* XL    = ws + off; off += (size_t)NTOT * D_GAT;      // 40.96M
  float* XR    = ws + off; off += (size_t)NTOT * D_GAT;      // 40.96M
  float* alpha = ws + off; off += (size_t)E_TOT * HEADS;     //  0.20M
  float* mbuf  = ws + off; off += (size_t)NTOT * HEADS;
  float* zbuf  = ws + off; off += (size_t)NTOT * HEADS;
  float* agg   = ws + off; off += (size_t)NTOT * D_EMB;      // 10.24M

  __bf16* bws = (__bf16*)(ws + off);
  size_t boff = 0;
  __bf16* xP   = bws + boff; boff += (size_t)N_X  * KB0 * 32;  // 3.74M
  __bf16* H1P  = bws + boff; boff += (size_t)N_X  * KB1 * 32;  // 4.61M
  __bf16* xcP  = bws + boff; boff += (size_t)NTOT * KB2 * 32;  // 10.24M
  __bf16* aggP = bws + boff; boff += (size_t)NTOT * KB2 * 32;  // 10.24M
  __bf16* W1P  = bws + boff; boff += (size_t)KB0 * D_H1  * 32;
  __bf16* W2P  = bws + boff; boff += (size_t)KB1 * D_EMB * 32;
  __bf16* WlP  = bws + boff; boff += (size_t)KB2 * D_GAT * 32;
  __bf16* WrP  = bws + boff; boff += (size_t)KB2 * D_GAT * 32;
  __bf16* WfcP = bws + boff; boff += (size_t)KB2 * D_OUT * 32;

  const int n_agg = NTOT * D_EMB;
  const int n_mz  = NTOT * HEADS;
  auto nb = [](size_t n) { return (unsigned)((n + 255) / 256); };
  auto blocks = [](int M, int N) { return dim3((N + 63) / 64, (M + 255) / 256); };

  // init accumulators / centroid rows of xc
  init_kernel<<<nb(n_agg), 256, 0, stream>>>(agg, mbuf, zbuf, n_agg, n_mz);
  hipMemcpyAsync(xc, emb, (size_t)N_C * D_EMB * sizeof(float),
                 hipMemcpyDeviceToDevice, stream);

  // pack all weights (one elementwise pass each)
  pack_B_kernel<<<nb((size_t)KB0 * D_H1 * 32),  256, 0, stream>>>(W1,  W1P,  D_IN0, D_H1,  KB0);
  pack_B_kernel<<<nb((size_t)KB1 * D_EMB * 32), 256, 0, stream>>>(W2,  W2P,  D_H1,  D_EMB, KB1);
  pack_B_kernel<<<nb((size_t)KB2 * D_GAT * 32), 256, 0, stream>>>(Wl,  WlP,  D_EMB, D_GAT, KB2);
  pack_B_kernel<<<nb((size_t)KB2 * D_GAT * 32), 256, 0, stream>>>(Wr,  WrP,  D_EMB, D_GAT, KB2);
  pack_B_kernel<<<nb((size_t)KB2 * D_OUT * 32), 256, 0, stream>>>(Wfc, WfcP, D_EMB, D_OUT, KB2);

  // MLP
  pack_A_kernel<<<nb((size_t)N_X * KB0 * 32), 256, 0, stream>>>(x, xP, N_X, D_IN0, KB0);
  gemm_wmma_packed<D_H1, true><<<blocks(N_X, D_H1), 256, 0, stream>>>(
      xP, W1P, b1, H1, N_X, KB0);
  pack_A_kernel<<<nb((size_t)N_X * KB1 * 32), 256, 0, stream>>>(H1, H1P, N_X, D_H1, KB1);
  gemm_wmma_packed<D_EMB, false><<<blocks(N_X, D_EMB), 256, 0, stream>>>(
      H1P, W2P, b2, xc + (size_t)N_C * D_EMB, N_X, KB1);

  // GAT projections (dominant GEMMs: 2 x 84 GFLOP)
  pack_A_kernel<<<nb((size_t)NTOT * KB2 * 32), 256, 0, stream>>>(xc, xcP, NTOT, D_EMB, KB2);
  gemm_wmma_packed<D_GAT, false><<<blocks(NTOT, D_GAT), 256, 0, stream>>>(
      xcP, WlP, bl, XL, NTOT, KB2);
  gemm_wmma_packed<D_GAT, false><<<blocks(NTOT, D_GAT), 256, 0, stream>>>(
      xcP, WrP, br, XR, NTOT, KB2);

  // attention scores + segment softmax
  const int n_eh = E_TOT * HEADS;
  alpha_kernel<<<(n_eh + 7) / 8, 256, 0, stream>>>(XL, XR, eidx, att, alpha);
  segmax_kernel<<<nb(n_eh), 256, 0, stream>>>(alpha, eidx, mbuf);
  expsum_kernel<<<nb(n_eh), 256, 0, stream>>>(alpha, eidx, mbuf, zbuf);

  // weighted scatter-aggregate (head-mean folded into coefficient)
  agg_kernel<<<E_TOT, 256, 0, stream>>>(XL, alpha, zbuf, eidx, agg);
  bias_add_kernel<<<nb(n_agg), 256, 0, stream>>>(agg, gatb, n_agg);

  // final FC -> first output, then pass-through exps
  pack_A_kernel<<<nb((size_t)NTOT * KB2 * 32), 256, 0, stream>>>(agg, aggP, NTOT, D_EMB, KB2);
  gemm_wmma_packed<D_OUT, false><<<blocks(NTOT, D_OUT), 256, 0, stream>>>(
      aggP, WfcP, bfc, out, NTOT, KB2);
  hipMemcpyAsync(out + (size_t)NTOT * D_OUT, exps,
                 (size_t)N_X * D_OUT * sizeof(float),
                 hipMemcpyDeviceToDevice, stream);
}
